// MultiHeadAttention_60000693125780
// MI455X (gfx1250) — matrix operand order repaired and checked by reading, not compiled, not run on hardware
//
#include <hip/hip_runtime.h>
#include <hip/hip_bf16.h>

// MI455X / gfx1250 wave32 WMMA implementation, v3.
// v2 + 2x2 register blocking: one wave computes a 32x32 output tile with
// 4 independent f32 accumulators -> 12 WMMAs per 16 b128 loads per K-chunk,
// and 4 interleaved accumulator chains to hide matrix-pipe latency.
// All GEMM operands are pre-packed f16 hi/lo planes (hi+lo product
// decomposition ~= fp32 accuracy); conv boundaries use a zero halo.

typedef __attribute__((ext_vector_type(16))) _Float16 v16h;
typedef __attribute__((ext_vector_type(8)))  _Float16 v8h;
typedef __attribute__((ext_vector_type(8)))  float    v8f;

#define WMMA(A, B, C) __builtin_amdgcn_wmma_f32_16x16x32_f16(false, (A), false, (B), (short)0, (C), false, false)

__device__ __forceinline__ v16h ldfrag(const _Float16* p0, const _Float16* p1) {
    v8h a = *(const v8h*)p0;
    v8h b = *(const v8h*)p1;
    return __builtin_shufflevector(a, b, 0, 1, 2, 3, 4, 5, 6, 7, 8, 9, 10, 11, 12, 13, 14, 15);
}

__device__ __forceinline__ void store_split(float v, _Float16* hi, _Float16* lo, int idx) {
    _Float16 h = (_Float16)v;
    hi[idx] = h;
    lo[idx] = (_Float16)(v - (float)h);
}

// Issue the 12 interleaved WMMAs of one 2x2-blocked hi/lo K-chunk.
#define CHUNK12()                         \
    acc00 = WMMA(a0h, b0h, acc00);        \
    acc01 = WMMA(a0h, b1h, acc01);        \
    acc10 = WMMA(a1h, b0h, acc10);        \
    acc11 = WMMA(a1h, b1h, acc11);        \
    acc00 = WMMA(a0h, b0l, acc00);        \
    acc01 = WMMA(a0h, b1l, acc01);        \
    acc10 = WMMA(a1h, b0l, acc10);        \
    acc11 = WMMA(a1h, b1l, acc11);        \
    acc00 = WMMA(a0l, b0h, acc00);        \
    acc01 = WMMA(a0l, b1h, acc01);        \
    acc10 = WMMA(a1l, b0h, acc10);        \
    acc11 = WMMA(a1l, b1h, acc11)

// ---------------------------------------------------------------------------
// NCHW f32 (n,512,HH,WW) -> padded NHWC f16 hi/lo planes (n, HH+2, WW+2, 512)
// ---------------------------------------------------------------------------
__global__ void mha_pack_input(const float* __restrict__ src, _Float16* __restrict__ dhi,
                               _Float16* __restrict__ dlo, int HH, int WW) {
    const int idx = blockIdx.x * 256 + threadIdx.x;
    const int ci = idx & 511;
    int r = idx >> 9;
    const int PW = WW + 2, PH = HH + 2;
    const int xp = r % PW; r /= PW;
    const int yp = r % PH;
    const int n  = r / PH;
    float v = 0.0f;
    if (yp >= 1 && yp <= HH && xp >= 1 && xp <= WW)
        v = src[((n * 512 + ci) * HH + (yp - 1)) * WW + (xp - 1)];
    store_split(v, dhi, dlo, idx);
}

// ---------------------------------------------------------------------------
// W (co,ci,3,3) f32 -> [kk][co][ci] f16 hi/lo planes (GEMM A layout).
// ---------------------------------------------------------------------------
__global__ void mha_pack_weight(const float* __restrict__ W, _Float16* __restrict__ dhi,
                                _Float16* __restrict__ dlo) {
    const int idx = blockIdx.x * 256 + threadIdx.x;  // (kk*512 + co)*512 + ci
    const int ci = idx & 511;
    const int co = (idx >> 9) & 511;
    const int kk = idx >> 18;
    store_split(W[(co * 512 + ci) * 9 + kk], dhi, dlo, idx);
}

__global__ void mha_fill_zero(unsigned int* __restrict__ p) {
    p[blockIdx.x * 256 + threadIdx.x] = 0u;
}

// ---------------------------------------------------------------------------
// Conv 3x3 'same' (Cin=Cout=512) as implicit GEMM; 32(co) x 32(pos) per wave.
// mode 0: scatter -> Q/K attention layout  [n2][d2][h'*32+r2]  (f16 hi/lo)
// mode 2: scatter -> V transposed layout   [n2][h'*32+r2][d2]  (f16 hi/lo)
// mode 1: write f32 row-major (n, co, HH, WW) to d_out
// ---------------------------------------------------------------------------
__global__ void mha_conv_gemm(const _Float16* __restrict__ Xhi, const _Float16* __restrict__ Xlo,
                              const _Float16* __restrict__ Whi, const _Float16* __restrict__ Wlo,
                              const float* __restrict__ bias,
                              _Float16* __restrict__ Ohi, _Float16* __restrict__ Olo,
                              float* __restrict__ Of32,
                              int HH, int WW, int tiles_per_img, int ptiles, int mode) {
    const int lane = threadIdx.x & 31;
    const int wave = threadIdx.x >> 5;
    const int tile = blockIdx.x * 4 + wave;
    const int n    = tile / tiles_per_img;
    const int t    = tile - n * tiles_per_img;
    const int cot  = t / ptiles;
    const int pt   = t - cot * ptiles;
    const int co0  = cot * 32;
    const int hh   = lane >> 4;
    const int ln   = lane & 15;
    const int PW   = WW + 2;

    int py[2], px[2];
#pragma unroll
    for (int b = 0; b < 2; ++b) {
        const int p = pt * 32 + b * 16 + ln;
        py[b] = p / WW;
        px[b] = p - py[b] * WW;
    }

    v8f acc00 = {}, acc01 = {}, acc10 = {}, acc11 = {};
    for (int kk = 0; kk < 9; ++kk) {
        const int dy = kk / 3, dx = kk - (kk / 3) * 3;        // padded offsets 0..2
        const _Float16* Ah0 = Whi + (kk * 512 + co0 + ln) * 512;
        const _Float16* Al0 = Wlo + (kk * 512 + co0 + ln) * 512;
        const _Float16* Ah1 = Ah0 + 16 * 512;
        const _Float16* Al1 = Al0 + 16 * 512;
        const int pix0 = ((n * (HH + 2) + py[0] + dy) * PW + (px[0] + dx)) * 512;
        const int pix1 = ((n * (HH + 2) + py[1] + dy) * PW + (px[1] + dx)) * 512;
        for (int c0 = 0; c0 < 512; c0 += 32) {
            v16h a0h = ldfrag(Ah0 + c0 + hh * 8,  Ah0 + c0 + 16 + hh * 8);
            v16h a0l = ldfrag(Al0 + c0 + hh * 8,  Al0 + c0 + 16 + hh * 8);
            v16h a1h = ldfrag(Ah1 + c0 + hh * 8,  Ah1 + c0 + 16 + hh * 8);
            v16h a1l = ldfrag(Al1 + c0 + hh * 8,  Al1 + c0 + 16 + hh * 8);
            v16h b0h = ldfrag(Xhi + pix0 + c0 + hh * 8, Xhi + pix0 + c0 + 16 + hh * 8);
            v16h b0l = ldfrag(Xlo + pix0 + c0 + hh * 8, Xlo + pix0 + c0 + 16 + hh * 8);
            v16h b1h = ldfrag(Xhi + pix1 + c0 + hh * 8, Xhi + pix1 + c0 + 16 + hh * 8);
            v16h b1l = ldfrag(Xlo + pix1 + c0 + hh * 8, Xlo + pix1 + c0 + 16 + hh * 8);
            CHUNK12();
        }
    }
#pragma unroll
    for (int a = 0; a < 2; ++a) {
#pragma unroll
        for (int b = 0; b < 2; ++b) {
            const v8f accv = (a == 0) ? (b == 0 ? acc00 : acc01)
                                      : (b == 0 ? acc10 : acc11);
#pragma unroll
            for (int j = 0; j < 8; ++j) {
                const int co = co0 + a * 16 + hh * 8 + j;
                const float val = accv[j] + bias[co];
                if (mode == 1) {
                    Of32[((n * 512 + co) * HH + py[b]) * WW + px[b]] = val;
                } else {
                    // (n, d=co, r=py, c=px) -> scrambled attention coords
                    const int n2 = (n & 3) * 8 + (co >> 6);
                    const int d2 = (co & 63) * 8 + (py[b] >> 2);
                    const int r2 = (py[b] & 3) * 8 + px[b];
                    const int hp = n >> 2;
                    const int idx = (mode == 0) ? (n2 * 512 + d2) * 256 + hp * 32 + r2
                                                : (n2 * 256 + hp * 32 + r2) * 512 + d2;
                    store_split(val, Ohi, Olo, idx);
                }
            }
        }
    }
}

// ---------------------------------------------------------------------------
// scores[n2][qd][kd] = sum_j Q[n2][qd][j]*K[n2][kd][j]  (K=256, scale=1)
// 32x32 tile per wave; written pre-softmax into the attn region of d_out.
// ---------------------------------------------------------------------------
__global__ void mha_scores(const _Float16* __restrict__ Qhi, const _Float16* __restrict__ Qlo,
                           const _Float16* __restrict__ Khi, const _Float16* __restrict__ Klo,
                           float* __restrict__ S) {
    const int lane = threadIdx.x & 31;
    const int wave = threadIdx.x >> 5;
    const int tile = blockIdx.x * 4 + wave;   // 32 * 16 * 16 tiles
    const int n2 = tile >> 8;
    const int t  = tile & 255;
    const int mt = t >> 4, nt = t & 15;
    const int hh = lane >> 4, ln = lane & 15;
    const int ar0 = (n2 * 512 + mt * 32 + ln) * 256;
    const int ar1 = ar0 + 16 * 256;
    const int br0 = (n2 * 512 + nt * 32 + ln) * 256;
    const int br1 = br0 + 16 * 256;

    v8f acc00 = {}, acc01 = {}, acc10 = {}, acc11 = {};
    for (int k0 = 0; k0 < 256; k0 += 32) {
        v16h a0h = ldfrag(Qhi + ar0 + k0 + hh * 8,  Qhi + ar0 + k0 + 16 + hh * 8);
        v16h a0l = ldfrag(Qlo + ar0 + k0 + hh * 8,  Qlo + ar0 + k0 + 16 + hh * 8);
        v16h a1h = ldfrag(Qhi + ar1 + k0 + hh * 8,  Qhi + ar1 + k0 + 16 + hh * 8);
        v16h a1l = ldfrag(Qlo + ar1 + k0 + hh * 8,  Qlo + ar1 + k0 + 16 + hh * 8);
        v16h b0h = ldfrag(Khi + br0 + k0 + hh * 8, Khi + br0 + k0 + 16 + hh * 8);
        v16h b0l = ldfrag(Klo + br0 + k0 + hh * 8, Klo + br0 + k0 + 16 + hh * 8);
        v16h b1h = ldfrag(Khi + br1 + k0 + hh * 8, Khi + br1 + k0 + 16 + hh * 8);
        v16h b1l = ldfrag(Klo + br1 + k0 + hh * 8, Klo + br1 + k0 + 16 + hh * 8);
        CHUNK12();
    }
#pragma unroll
    for (int a = 0; a < 2; ++a) {
#pragma unroll
        for (int b = 0; b < 2; ++b) {
            const v8f accv = (a == 0) ? (b == 0 ? acc00 : acc01)
                                      : (b == 0 ? acc10 : acc11);
#pragma unroll
            for (int j = 0; j < 8; ++j) {
                const int qd = mt * 32 + a * 16 + hh * 8 + j;
                const int kd = nt * 32 + b * 16 + ln;
                S[(n2 * 512 + qd) * 512 + kd] = accv[j];
            }
        }
    }
}

// ---------------------------------------------------------------------------
// In-place row softmax over 512 keys; also emits hi/lo f16 planes of P.
// ---------------------------------------------------------------------------
__global__ void mha_softmax(float* __restrict__ A, _Float16* __restrict__ Phi,
                            _Float16* __restrict__ Plo) {
    const int lane = threadIdx.x & 31;
    const int wave = threadIdx.x >> 5;
    const int row  = blockIdx.x * 4 + wave;   // 32*512 rows
    float* p = A + row * 512;
    float v[16];
    float mx = -3.4e38f;
#pragma unroll
    for (int i = 0; i < 16; ++i) { v[i] = p[lane + i * 32]; mx = fmaxf(mx, v[i]); }
#pragma unroll
    for (int off = 16; off > 0; off >>= 1) mx = fmaxf(mx, __shfl_xor(mx, off, 32));
    float sum = 0.0f;
#pragma unroll
    for (int i = 0; i < 16; ++i) { v[i] = expf(v[i] - mx); sum += v[i]; }
#pragma unroll
    for (int off = 16; off > 0; off >>= 1) sum += __shfl_xor(sum, off, 32);
    const float inv = 1.0f / sum;
#pragma unroll
    for (int i = 0; i < 16; ++i) {
        const float pv = v[i] * inv;
        const int idx = row * 512 + lane + i * 32;
        p[lane + i * 32] = pv;
        store_split(pv, Phi, Plo, idx);
    }
}

// ---------------------------------------------------------------------------
// O[n2][qd][j] = sum_kd P[n2][qd][kd]*V[n2][kd][j]  (K=512), 32x32 per wave.
// Scatter into padded NHWC image (4, 10, 258, 512) of the final conv.
// ---------------------------------------------------------------------------
__global__ void mha_pv(const _Float16* __restrict__ Phi, const _Float16* __restrict__ Plo,
                       const _Float16* __restrict__ Vthi, const _Float16* __restrict__ Vtlo,
                       _Float16* __restrict__ X2hi, _Float16* __restrict__ X2lo) {
    const int lane = threadIdx.x & 31;
    const int wave = threadIdx.x >> 5;
    const int tile = blockIdx.x * 4 + wave;   // 32 * 16 * 8 tiles
    const int n2 = tile >> 7;
    const int t  = tile & 127;
    const int mt = t >> 3, nt = t & 7;
    const int hh = lane >> 4, ln = lane & 15;
    const int ar0 = (n2 * 512 + mt * 32 + ln) * 512;
    const int ar1 = ar0 + 16 * 512;
    const int br0 = (n2 * 256 + nt * 32 + ln) * 512;
    const int br1 = br0 + 16 * 512;

    v8f acc00 = {}, acc01 = {}, acc10 = {}, acc11 = {};
    for (int k0 = 0; k0 < 512; k0 += 32) {
        v16h a0h = ldfrag(Phi + ar0 + k0 + hh * 8,   Phi + ar0 + k0 + 16 + hh * 8);
        v16h a0l = ldfrag(Plo + ar0 + k0 + hh * 8,   Plo + ar0 + k0 + 16 + hh * 8);
        v16h a1h = ldfrag(Phi + ar1 + k0 + hh * 8,   Phi + ar1 + k0 + 16 + hh * 8);
        v16h a1l = ldfrag(Plo + ar1 + k0 + hh * 8,   Plo + ar1 + k0 + 16 + hh * 8);
        v16h b0h = ldfrag(Vthi + br0 + k0 + hh * 8, Vthi + br0 + k0 + 16 + hh * 8);
        v16h b0l = ldfrag(Vtlo + br0 + k0 + hh * 8, Vtlo + br0 + k0 + 16 + hh * 8);
        v16h b1h = ldfrag(Vthi + br1 + k0 + hh * 8, Vthi + br1 + k0 + 16 + hh * 8);
        v16h b1l = ldfrag(Vtlo + br1 + k0 + hh * 8, Vtlo + br1 + k0 + 16 + hh * 8);
        CHUNK12();
    }
#pragma unroll
    for (int a = 0; a < 2; ++a) {
#pragma unroll
        for (int b = 0; b < 2; ++b) {
            const v8f accv = (a == 0) ? (b == 0 ? acc00 : acc01)
                                      : (b == 0 ? acc10 : acc11);
#pragma unroll
            for (int j = 0; j < 8; ++j) {
                const int qd = mt * 32 + a * 16 + hh * 8 + j;
                const int jc = nt * 32 + b * 16 + ln;
                const int hatt = jc >> 5, r2 = jc & 31;
                const int bb = n2 >> 3, s = n2 & 7;
                const int d = (qd & 1) * 256 + r2 * 8 + hatt;
                const int w = (qd >> 4) * 8 + ((qd >> 1) & 7);
                const int idx = ((bb * 10 + s + 1) * 258 + (w + 1)) * 512 + d; // padded NHWC
                store_split(accv[j], X2hi, X2lo, idx);
            }
        }
    }
}

// ---------------------------------------------------------------------------
extern "C" void kernel_launch(void* const* d_in, const int* in_sizes, int n_in,
                              void* d_out, int out_size, void* d_ws, size_t ws_size,
                              hipStream_t stream) {
    const float* q  = (const float*)d_in[0];
    const float* k  = (const float*)d_in[1];
    const float* v  = (const float*)d_in[2];
    const float* Wq = (const float*)d_in[3];
    const float* bq = (const float*)d_in[4];
    const float* Wk = (const float*)d_in[5];
    const float* bk = (const float*)d_in[6];
    const float* Wv = (const float*)d_in[7];
    const float* bv = (const float*)d_in[8];
    const float* Wo = (const float*)d_in[9];
    const float* bo = (const float*)d_in[10];

    float* y    = (float*)d_out;                 // (4,512,8,32,8) = 4,194,304 f32
    float* attn = y + 4194304;                   // (32,512,512)   = 8,388,608 f32

    // Workspace layout (units: _Float16). Total ~94.3 MB.
    _Float16* ws = (_Float16*)d_ws;
    _Float16* Qhi = ws;                          // 4,194,304 each
    _Float16* Qlo = ws + 4194304;
    _Float16* Khi = ws + 8388608;
    _Float16* Klo = ws + 12582912;
    _Float16* Vthi = ws + 16777216;
    _Float16* Vtlo = ws + 20971520;
    _Float16* Phi = ws + 25165824;
    _Float16* Plo = ws + 29360128;
    _Float16* XPhi = ws + 33554432;              // 5,570,560 (q/k/v padded NHWC; reused for X2)
    _Float16* XPlo = ws + 39124992;
    _Float16* WRhi = ws + 44695552;              // 2,359,296 (reused per weight tensor)
    _Float16* WRlo = ws + 47054848;

    const dim3 b256(256), b128(128);

    // ---- Q projection ----
    mha_pack_weight<<<9216, b256, 0, stream>>>(Wq, WRhi, WRlo);
    mha_pack_input<<<21760, b256, 0, stream>>>(q, XPhi, XPlo, 32, 8);
    mha_conv_gemm<<<1024, b128, 0, stream>>>(XPhi, XPlo, WRhi, WRlo, bq, Qhi, Qlo, nullptr,
                                             32, 8, 128, 8, 0);
    // ---- K projection ----
    mha_pack_weight<<<9216, b256, 0, stream>>>(Wk, WRhi, WRlo);
    mha_pack_input<<<21760, b256, 0, stream>>>(k, XPhi, XPlo, 32, 8);
    mha_conv_gemm<<<1024, b128, 0, stream>>>(XPhi, XPlo, WRhi, WRlo, bk, Khi, Klo, nullptr,
                                             32, 8, 128, 8, 0);
    // ---- V projection (transposed layout) ----
    mha_pack_weight<<<9216, b256, 0, stream>>>(Wv, WRhi, WRlo);
    mha_pack_input<<<21760, b256, 0, stream>>>(v, XPhi, XPlo, 32, 8);
    mha_conv_gemm<<<1024, b128, 0, stream>>>(XPhi, XPlo, WRhi, WRlo, bv, Vthi, Vtlo, nullptr,
                                             32, 8, 128, 8, 2);
    // ---- scores + softmax (attn written to d_out) ----
    mha_scores<<<2048, b128, 0, stream>>>(Qhi, Qlo, Khi, Klo, attn);
    mha_softmax<<<4096, b128, 0, stream>>>(attn, Phi, Plo);
    // ---- P@V into zero-filled padded NHWC image ----
    mha_fill_zero<<<21760, b256, 0, stream>>>((unsigned int*)XPhi);  // clears XPhi+XPlo region
    mha_pv<<<1024, b128, 0, stream>>>(Phi, Plo, Vthi, Vtlo, XPhi, XPlo);
    // ---- output conv: (4, 10, 258, 512) padded NHWC -> y (4,512,8,256) f32 ----
    mha_pack_weight<<<9216, b256, 0, stream>>>(Wo, WRhi, WRlo);
    mha_conv_gemm<<<1024, b128, 0, stream>>>(XPhi, XPlo, WRhi, WRlo, bo, nullptr, nullptr, y,
                                             8, 256, 1024, 64, 1);
}
